// VAE_35553739277006
// MI455X (gfx1250) — matrix operand order repaired and checked by reading, not compiled, not run on hardware
//
#include <hip/hip_runtime.h>
#include <hip/hip_bf16.h>
#include <math.h>

typedef __attribute__((ext_vector_type(16))) _Float16 v16h;
typedef __attribute__((ext_vector_type(8)))  float    v8f;

#define B_SAMP   8192
#define D_INC    12288
#define HID      300
#define HIDP     320      // HID padded to multiple of 32 (f16 activation row stride)
#define NODEC    8
#define VMIN_C   (-5.0f)
#define INT_LEN_C (10.0f / 99.0f)

constexpr int BM = 128, BN = 64, BK = 32;
constexpr int LDT = BK + 8;   // LDS row stride in halfs (80B, 16B-aligned, bank-conflict pad)

union F16Frag { unsigned u[8]; v16h v; };
union H16     { _Float16 h[16]; uint4 q[2]; };

// ---------------------------------------------------------------------------
// Tiled WMMA GEMM:  C[M,N] = act(A[M,K] @ B[K,N] + bias[N])
//   ACT: 0=none, 1=ELU, 2=tanh
//   AH:  A is f16 with padded row stride KpA (zero-padded cols, so any K works)
//        else A is f32 with row stride K  -- REQUIRES K % 32 == 0
//   OUTH: write f16 output with row stride ldC, else f32 with row stride ldC
// B is pre-transposed/pre-padded f16: Bt[n][k], row stride KpB (mult of 32),
// rows padded to a multiple of BN, zero-filled -> staging is guard-free.
// M must be a multiple of 128.
// Software-pipelined: tile k+1 is fetched to registers while tile k computes.
// ---------------------------------------------------------------------------
template <int ACT, bool AH, bool OUTH>
__global__ __launch_bounds__(256) void gemm_wmma_kernel(
    const void* __restrict__ Aptr, const _Float16* __restrict__ Bt,
    const float* __restrict__ bias, void* __restrict__ Cptr,
    int M, int K, int N, int KpA, int KpB, int ldC) {
  __shared__ __align__(16) _Float16 As[BM][LDT];
  __shared__ __align__(16) _Float16 Bs[BN][LDT];   // transposed: Bs[n][k]

  const int tid    = threadIdx.x;
  const int lane   = tid & 31;
  const int wid    = tid >> 5;
  const int wave_m = wid & 3;       // 4 waves along M
  const int wave_n = wid >> 2;      // 2 waves along N
  const int lsel   = lane >> 4;     // 0: lanes 0-15, 1: lanes 16-31
  const int lmod   = lane & 15;
  const int m0     = blockIdx.y * BM;
  const int n0     = blockIdx.x * BN;

  // A-tile loader: 2 threads per row, 16 consecutive K each
  const int ar  = tid >> 1;
  const int ac0 = (tid & 1) * 16;
  // B-tile loader: one N row of Bt per thread, 8 consecutive K
  const int bn  = tid & 63;
  const int bk0 = (tid >> 6) * 8;

  const _Float16* __restrict__ Ah = (const _Float16*)Aptr;
  const float*    __restrict__ Af = (const float*)Aptr;
  const _Float16* brow = Bt + (long)(n0 + bn) * KpB + bk0;

  v8f acc[2][2] = {};

  // staging registers (double-buffer in regs, single LDS buffer)
  uint4  ra0, ra1, rb;      // AH path + B
  float4 rf0, rf1, rf2, rf3; // f32-A path raw data (convert at store time)

  auto fetch = [&](int k0) {
    if constexpr (AH) {
      const _Float16* arow = Ah + (long)(m0 + ar) * KpA + k0 + ac0;
      ra0 = ((const uint4*)arow)[0];
      ra1 = ((const uint4*)arow)[1];
    } else {
      const float4* p = (const float4*)(Af + (long)(m0 + ar) * K + k0 + ac0);
      rf0 = p[0]; rf1 = p[1]; rf2 = p[2]; rf3 = p[3];
    }
    rb = *(const uint4*)(brow + k0);
  };

  fetch(0);

  for (int k0 = 0; k0 < K; k0 += BK) {
    // ---- commit staged tile to LDS ----
    if constexpr (AH) {
      *(uint4*)&As[ar][ac0]     = ra0;
      *(uint4*)&As[ar][ac0 + 8] = ra1;
    } else {
      H16 hh;
      hh.h[0]  = (_Float16)rf0.x; hh.h[1]  = (_Float16)rf0.y;
      hh.h[2]  = (_Float16)rf0.z; hh.h[3]  = (_Float16)rf0.w;
      hh.h[4]  = (_Float16)rf1.x; hh.h[5]  = (_Float16)rf1.y;
      hh.h[6]  = (_Float16)rf1.z; hh.h[7]  = (_Float16)rf1.w;
      hh.h[8]  = (_Float16)rf2.x; hh.h[9]  = (_Float16)rf2.y;
      hh.h[10] = (_Float16)rf2.z; hh.h[11] = (_Float16)rf2.w;
      hh.h[12] = (_Float16)rf3.x; hh.h[13] = (_Float16)rf3.y;
      hh.h[14] = (_Float16)rf3.z; hh.h[15] = (_Float16)rf3.w;
      *(uint4*)&As[ar][ac0]     = hh.q[0];
      *(uint4*)&As[ar][ac0 + 8] = hh.q[1];
    }
    *(uint4*)&Bs[bn][bk0] = rb;
    __syncthreads();

    // ---- issue next tile's global loads (overlap with WMMA below) ----
    const int k1 = k0 + BK;
    if (k1 < K) fetch(k1);

    // ---- fragments per the CDNA5 16-bit A/B VGPR layouts ----
    F16Frag fa[2], fb[2];
    #pragma unroll
    for (int s = 0; s < 2; ++s) {
      const int m = wave_m * 32 + s * 16 + lmod;
      #pragma unroll
      for (int h = 0; h < 8; ++h) {
        // A 16x32: lanes 0-15 hold K {0..7,16..23}; lanes 16-31 hold K {8..15,24..31}
        const int kp = ((h < 4) ? (2 * h) : (2 * h + 8)) + lsel * 8;
        fa[s].u[h] = *(const unsigned*)&As[m][kp];
      }
      const int n = wave_n * 32 + s * 16 + lmod;
      #pragma unroll
      for (int h = 0; h < 8; ++h) {
        // B 32x16: lanes 0-15 hold K {0..7,16..23}; lanes 16-31 hold K {8..15,24..31}
        const int kp = ((h < 4) ? (2 * h) : (2 * h + 8)) + lsel * 8;
        fb[s].u[h] = *(const unsigned*)&Bs[n][kp];
      }
    }

    #pragma unroll
    for (int sm = 0; sm < 2; ++sm)
      #pragma unroll
      for (int sn = 0; sn < 2; ++sn)
        acc[sm][sn] = __builtin_amdgcn_wmma_f32_16x16x32_f16(
            false, fa[sm].v, false, fb[sn].v, (short)0, acc[sm][sn], false, false);
    __syncthreads();
  }

  // epilogue: bias + activation (D layout: vgpr r -> M=r(+8), lane -> N)
  #pragma unroll
  for (int sm = 0; sm < 2; ++sm) {
    #pragma unroll
    for (int sn = 0; sn < 2; ++sn) {
      const int n = n0 + wave_n * 32 + sn * 16 + lmod;
      if (n < N) {
        const float bv = bias[n];
        #pragma unroll
        for (int r = 0; r < 8; ++r) {
          const int m = m0 + wave_m * 32 + sm * 16 + r + lsel * 8;
          float v = acc[sm][sn][r] + bv;
          if (ACT == 1)      v = (v > 0.0f) ? v : (expf(v) - 1.0f);
          else if (ACT == 2) v = tanhf(v);
          if constexpr (OUTH) ((_Float16*)Cptr)[(long)m * ldC + n] = (_Float16)v;
          else                ((float*)Cptr)[(long)m * ldC + n]    = v;
        }
      }
    }
  }
}

// ---------------------------------------------------------------------------
// One-time weight prep: transpose K x N f32 -> Np x Kp f16 (zero padded).
// ---------------------------------------------------------------------------
__global__ void transpose_w_kernel(const float* __restrict__ Wsrc, _Float16* __restrict__ Bt,
                                   int K, int N, int Kp, int Np) {
  const long idx = (long)blockIdx.x * blockDim.x + threadIdx.x;
  if (idx >= (long)Np * Kp) return;
  const int n = (int)(idx / Kp);
  const int k = (int)(idx % Kp);
  const float v = (n < N && k < K) ? Wsrc[(long)k * N + n] : 0.0f;
  Bt[idx] = (_Float16)v;
}

// zero the padded columns [Kc, Kp) of an f16 activation buffer
__global__ void padzero_kernel(_Float16* __restrict__ buf, int rows, int Kc, int Kp) {
  const int w   = Kp - Kc;
  const int idx = blockIdx.x * blockDim.x + threadIdx.x;
  if (idx >= rows * w) return;
  const int r = idx / w, c = idx % w;
  buf[(long)r * Kp + Kc + c] = (_Float16)0.0f;
}

// ---------------------------------------------------------------------------
// 8x8 (I - W*mask)^-1 (unit lower-triangular, Gauss-Jordan) + PWL tables.
// ---------------------------------------------------------------------------
__global__ void prep_kernel(const float* __restrict__ W, const float* __restrict__ mask,
                            const float* __restrict__ pwl_p, const float* __restrict__ pwl_b,
                            float* __restrict__ inv, float* __restrict__ ew,
                            float* __restrict__ db) {
  const int t = threadIdx.x;
  if (t == 0) {
    float Am[8][16];
    for (int i = 0; i < 8; ++i)
      for (int j = 0; j < 8; ++j) {
        Am[i][j]     = ((i == j) ? 1.0f : 0.0f) - W[i * 8 + j] * mask[i * 8 + j];
        Am[i][8 + j] = (i == j) ? 1.0f : 0.0f;
      }
    for (int p = 0; p < 8; ++p) {
      const float r = 1.0f / Am[p][p];
      for (int j = 0; j < 16; ++j) Am[p][j] *= r;
      for (int i = 0; i < 8; ++i)
        if (i != p) {
          const float f = Am[i][p];
          for (int j = 0; j < 16; ++j) Am[i][j] -= f * Am[p][j];
        }
    }
    for (int i = 0; i < 8; ++i)
      for (int j = 0; j < 8; ++j) inv[i * 8 + j] = Am[i][8 + j];
  }
  if (t < NODEC) {
    for (int i = 0; i <= 100; ++i) ew[t * 101 + i] = expf(pwl_p[t * 101 + i]) + 0.001f;
    float acc = pwl_b[t];
    db[t * 100 + 0] = acc;
    for (int i = 1; i < 100; ++i) {
      acc += INT_LEN_C * (expf(pwl_p[t * 101 + i]) + 0.001f);
      db[t * 100 + i] = acc;
    }
  }
}

__device__ __forceinline__ float pwl_apply(float eps, const float* __restrict__ ew,
                                           const float* __restrict__ db) {
  int index = (int)floorf((eps - VMIN_C) / INT_LEN_C) + 1;
  index = (index < 0) ? 0 : ((index > 100) ? 100 : index);
  int start = index - 1;
  if (start < 0) start = 0;
  const float sp = VMIN_C + (float)start * INT_LEN_C;
  return (eps - sp) * ew[index] + db[start];
}

// ---------------------------------------------------------------------------
// Per-sample latent path (reads f32 H3 of shape [B,16]).
// ---------------------------------------------------------------------------
__global__ void latent_kernel(const float* __restrict__ H3, const float* __restrict__ noise,
                              const float* __restrict__ inv, const float* __restrict__ ew,
                              const float* __restrict__ db, float* __restrict__ out) {
  const int b = blockIdx.x * blockDim.x + threadIdx.x;
  if (b >= B_SAMP) return;
  float mean[8], lv[8], eps[8];
  #pragma unroll
  for (int k = 0; k < 8; ++k) {
    mean[k] = H3[b * 16 + k];
    lv[k]   = H3[b * 16 + 8 + k];
    eps[k]  = mean[k] + expf(0.5f * lv[k]) * noise[b * 8 + k];
  }
  float* o_mean  = out;
  float* o_lv    = out + 1 * B_SAMP * 8;
  float* o_orig  = out + 2 * B_SAMP * 8;
  float* o_lat   = out + 3 * B_SAMP * 8;
  float* o_align = out + 4 * B_SAMP * 8;
  #pragma unroll
  for (int j = 0; j < 8; ++j) {
    float s = 0.0f, s2 = 0.0f;
    #pragma unroll
    for (int k = 0; k < 8; ++k) {
      const float iv = inv[k * 8 + j];
      s  += eps[k] * iv;
      s2 += mean[k] * iv;
    }
    o_mean[b * 8 + j]  = mean[j];
    o_lv[b * 8 + j]    = lv[j];
    o_orig[b * 8 + j]  = s;
    o_lat[b * 8 + j]   = pwl_apply(s,  ew + j * 101, db + j * 100);
    o_align[b * 8 + j] = pwl_apply(s2, ew + j * 101, db + j * 100);
  }
}

// decoder layer 1: K=8 dot per output, ELU, write f16 with zero padding to HIDP
__global__ void dec1_kernel(const float* __restrict__ latent, const float* __restrict__ w1,
                            const float* __restrict__ b1, _Float16* __restrict__ D1h) {
  const int idx = blockIdx.x * blockDim.x + threadIdx.x;
  if (idx >= B_SAMP * HIDP) return;
  const int b = idx / HIDP, j = idx % HIDP;
  float s = 0.0f;
  if (j < HID) {
    s = b1[j];
    #pragma unroll
    for (int k = 0; k < 8; ++k) s += latent[b * 8 + k] * w1[k * HID + j];
    s = (s > 0.0f) ? s : (expf(s) - 1.0f);
  }
  D1h[idx] = (_Float16)s;
}

// ---------------------------------------------------------------------------
extern "C" void kernel_launch(void* const* d_in, const int* in_sizes, int n_in,
                              void* d_out, int out_size, void* d_ws, size_t ws_size,
                              hipStream_t stream) {
  const float* x      = (const float*)d_in[0];
  const float* noise  = (const float*)d_in[1];
  const float* enc_w1 = (const float*)d_in[2];
  const float* enc_b1 = (const float*)d_in[3];
  const float* enc_w2 = (const float*)d_in[4];
  const float* enc_b2 = (const float*)d_in[5];
  const float* enc_w3 = (const float*)d_in[6];
  const float* enc_b3 = (const float*)d_in[7];
  const float* dec_w1 = (const float*)d_in[8];
  const float* dec_b1 = (const float*)d_in[9];
  const float* dec_w2 = (const float*)d_in[10];
  const float* dec_b2 = (const float*)d_in[11];
  const float* dec_w3 = (const float*)d_in[12];
  const float* dec_b3 = (const float*)d_in[13];
  const float* W      = (const float*)d_in[14];
  const float* mask   = (const float*)d_in[15];
  const float* pwl_p  = (const float*)d_in[16];
  const float* pwl_b  = (const float*)d_in[17];

  float* out = (float*)d_out;
  float* ws  = (float*)d_ws;

  // f32 workspace
  float* inv = ws;            // 64
  float* ew  = ws + 64;       // 808
  float* db  = ws + 872;      // 800
  float* H3  = ws + 2048;     // 8192*16

  // f16 workspace (weights transposed+padded, activations padded to stride 320)
  _Float16* h16  = (_Float16*)(ws + 2048 + (size_t)B_SAMP * 16);
  _Float16* w1t  = h16;                                   // 320  x 12288
  _Float16* w2t  = w1t  + (size_t)HIDP * D_INC;           // 320  x 320
  _Float16* w3t  = w2t  + (size_t)HIDP * HIDP;            // 64   x 320
  _Float16* dw2t = w3t  + (size_t)64 * HIDP;              // 320  x 320
  _Float16* dw3t = dw2t + (size_t)HIDP * HIDP;            // 12288 x 320
  _Float16* H1h  = dw3t + (size_t)D_INC * HIDP;           // 8192 x 320
  _Float16* H2h  = H1h  + (size_t)B_SAMP * HIDP;
  _Float16* D1h  = H2h  + (size_t)B_SAMP * HIDP;
  _Float16* D2h  = D1h  + (size_t)B_SAMP * HIDP;

  float* o_latent = out + 3 * B_SAMP * 8;
  float* o_xhat   = out + 5 * B_SAMP * 8;

  // ---- one-time prep (runs each launch; deterministic) ----
  prep_kernel<<<1, 64, 0, stream>>>(W, mask, pwl_p, pwl_b, inv, ew, db);

  const int TB = 256;
  transpose_w_kernel<<<((long)HIDP * D_INC + TB - 1) / TB, TB, 0, stream>>>(enc_w1, w1t,  D_INC, HID,   D_INC, HIDP);
  transpose_w_kernel<<<(HIDP * HIDP + TB - 1) / TB,       TB, 0, stream>>>(enc_w2, w2t,  HID,   HID,   HIDP,  HIDP);
  transpose_w_kernel<<<(64 * HIDP + TB - 1) / TB,         TB, 0, stream>>>(enc_w3, w3t,  HID,   16,    HIDP,  64);
  transpose_w_kernel<<<(HIDP * HIDP + TB - 1) / TB,       TB, 0, stream>>>(dec_w2, dw2t, HID,   HID,   HIDP,  HIDP);
  transpose_w_kernel<<<((long)D_INC * HIDP + TB - 1) / TB, TB, 0, stream>>>(dec_w3, dw3t, HID,  D_INC, HIDP,  D_INC);

  padzero_kernel<<<(B_SAMP * (HIDP - HID) + TB - 1) / TB, TB, 0, stream>>>(H1h, B_SAMP, HID, HIDP);
  padzero_kernel<<<(B_SAMP * (HIDP - HID) + TB - 1) / TB, TB, 0, stream>>>(H2h, B_SAMP, HID, HIDP);
  padzero_kernel<<<(B_SAMP * (HIDP - HID) + TB - 1) / TB, TB, 0, stream>>>(D2h, B_SAMP, HID, HIDP);

  // ---- encoder ----
  gemm_wmma_kernel<1, false, true><<<dim3(5, 64), 256, 0, stream>>>(
      x, w1t, enc_b1, H1h, B_SAMP, D_INC, HID, 0, D_INC, HIDP);
  gemm_wmma_kernel<1, true, true><<<dim3(5, 64), 256, 0, stream>>>(
      H1h, w2t, enc_b2, H2h, B_SAMP, HID, HID, HIDP, HIDP, HIDP);
  gemm_wmma_kernel<0, true, false><<<dim3(1, 64), 256, 0, stream>>>(
      H2h, w3t, enc_b3, H3, B_SAMP, HID, 16, HIDP, HIDP, 16);

  // ---- latent path ----
  latent_kernel<<<B_SAMP / 256, 256, 0, stream>>>(H3, noise, inv, ew, db, out);

  // ---- decoder ----
  dec1_kernel<<<(B_SAMP * HIDP + TB - 1) / TB, TB, 0, stream>>>(o_latent, dec_w1, dec_b1, D1h);
  gemm_wmma_kernel<1, true, true><<<dim3(5, 64), 256, 0, stream>>>(
      D1h, dw2t, dec_b2, D2h, B_SAMP, HID, HID, HIDP, HIDP, HIDP);
  gemm_wmma_kernel<2, true, false><<<dim3(192, 64), 256, 0, stream>>>(
      D2h, dw3t, dec_b3, o_xhat, B_SAMP, HID, D_INC, HIDP, HIDP, D_INC);
}
